// MCInfoNCE_34763465294555
// MI455X (gfx1250) — matrix operand order repaired and checked by reading, not compiled, not run on hardware
//
#include <hip/hip_runtime.h>
#include <stdint.h>

#define B_N 1024
#define D_N 256
#define S_N 32
#define KPOSF 16.0f
#define TPAD_H 272   // padded B-tile row stride in halves (256 + 16) -> 544B, 16B aligned

#if __has_builtin(__builtin_amdgcn_sched_barrier)
#define SCHED_BARRIER() __builtin_amdgcn_sched_barrier(0)
#else
#define SCHED_BARRIER() do {} while (0)
#endif

typedef __attribute__((ext_vector_type(16))) _Float16 v16h;
typedef __attribute__((ext_vector_type(8)))  _Float16 v8h;
typedef __attribute__((ext_vector_type(8)))  float    v8f;

// ---------------- deterministic stateless RNG helpers ----------------
__device__ __forceinline__ uint32_t hashu(uint32_t x) {
  x ^= x >> 17; x *= 0xed5ad4bbu;
  x ^= x >> 11; x *= 0xac4c1b51u;
  x ^= x >> 15; x *= 0x31848babu;
  x ^= x >> 14;
  return x;
}
__device__ __forceinline__ float u01(uint32_t h) {
  return ((float)h + 0.5f) * 2.3283064365386963e-10f;   // (h+0.5)/2^32 in (0,1)
}
__device__ __forceinline__ float gauss(uint32_t h1, uint32_t h2) {
  float r = sqrtf(-2.0f * __logf(u01(h1)));
  return r * __cosf(6.28318530717958647f * u01(h2));
}

// block = 256 threads (8 waves of 32)
__device__ __forceinline__ float block_reduce_sum(float v, float* sh) {
  #pragma unroll
  for (int off = 16; off > 0; off >>= 1) v += __shfl_xor(v, off, 32);
  const int lane = threadIdx.x & 31, wv = threadIdx.x >> 5;
  if (lane == 0) sh[wv] = v;
  __syncthreads();
  if (threadIdx.x == 0) {
    float t = 0.0f;
    #pragma unroll
    for (int i = 0; i < 8; ++i) t += sh[i];
    sh[0] = t;
  }
  __syncthreads();
  float out = sh[0];
  __syncthreads();
  return out;
}

// ---------------- kernel 1: normalize mu, Householder vector, vMF consts ----
__global__ __launch_bounds__(256) void vmf_prep(
    const float* __restrict__ mu1, const float* __restrict__ k1r,
    const float* __restrict__ mu2, const float* __restrict__ k2r,
    float* __restrict__ un, float* __restrict__ cst)
{
  const int b = blockIdx.x, pair = blockIdx.y, t = threadIdx.x;
  __shared__ float sh[8];
  const float* mu = (pair ? mu2 : mu1) + (size_t)b * D_N;
  float x = mu[t];
  float nrm = sqrtf(fmaxf(block_reduce_sum(x * x, sh), 1e-30f));
  float mn = x / nrm;
  float up = (t == 0 ? 1.0f : 0.0f) - mn;
  float upn = sqrtf(fmaxf(block_reduce_sum(up * up, sh), 1e-14f));
  un[((size_t)pair * B_N + b) * D_N + t] = up / upn;
  if (t == 0) {
    float kraw = (pair ? k2r : k1r)[b];
    float kap  = 20.0f + 80.0f * kraw;
    float m1   = (float)(D_N - 1);
    float quad = sqrtf(4.0f * kap * kap + m1 * m1);
    float bt   = (-2.0f * kap + quad) / m1;
    float ba   = m1 / (4.0f * kap);
    float mix  = fminf(fmaxf(kap - 10.0f, 0.0f), 1.0f);
    float bb   = ba * mix + bt * (1.0f - mix);
    float aa   = (m1 + 2.0f * kap + quad) * 0.25f;
    float dd   = 4.0f * aa * bb / (1.0f + bb) - m1 * __logf(m1);
    ((float4*)cst)[pair * B_N + b] = make_float4(bb, aa, dd, kap);
  }
}

// ---------------- kernel 2: vMF rejection sampler + Householder ------------
__global__ __launch_bounds__(256) void vmf_sample(
    const float* __restrict__ un, const float* __restrict__ cst,
    _Float16* __restrict__ z1h, _Float16* __restrict__ z2h)
{
  const int b = blockIdx.x, s = blockIdx.y, pair = blockIdx.z;
  const int t = threadIdx.x;
  __shared__ float sh[8];
  __shared__ float shw;

  const float4 c4 = ((const float4*)cst)[pair * B_N + b];
  const uint32_t base = (((uint32_t)pair * S_N + s) * B_N + b) * D_N + t;

  if (t == 0) {
    const float bb = c4.x, aa = c4.y, dd = c4.z;
    float eps_sel = 0.5f;
    bool acc = false;
    #pragma unroll
    for (int p = 0; p < 8; ++p) {
      if (!acc) {
        uint32_t h1 = hashu(base ^ (0x9e3779b9u * (p + 1)));
        uint32_t h2 = hashu(h1 ^ 0x85ebca6bu);
        uint32_t h3 = hashu(h2 ^ 0xc2b2ae35u);
        // Beta(127.5,127.5) ~= N(0.5, 1/32^2) for large symmetric params
        float n   = gauss(h1, h2);
        float eps = fminf(fmaxf(0.5f + 0.03125f * n, 1e-6f), 1.0f - 1e-6f);
        float u   = u01(h3);
        float tt  = 2.0f * aa * bb / (1.0f - (1.0f - bb) * eps);
        eps_sel = eps;
        if (255.0f * __logf(tt) - tt + dd > __logf(u)) acc = true;
      }
    }
    shw = (1.0f - (1.0f + c4.x) * eps_sel) / (1.0f - (1.0f - c4.x) * eps_sel);
  }

  // tangential direction (t==0 slot is the w coordinate)
  uint32_t g1 = hashu(base ^ 0x68bc21ebu);
  uint32_t g2 = hashu(g1 ^ 0x02e5be93u);
  float vi = (t == 0) ? 0.0f : gauss(g1, g2);
  float vsum = block_reduce_sum(vi * vi, sh);     // syncthreads inside -> shw visible
  float w = shw;
  float wcos = sqrtf(fmaxf(1.0f - w * w, 1e-14f));
  float xi = (t == 0) ? w : wcos * vi * rsqrtf(fmaxf(vsum, 1e-30f));

  // Householder reflection e1 -> mu
  float u_i = un[((size_t)pair * B_N + b) * D_N + t];
  float dot = block_reduce_sum(xi * u_i, sh);
  float zi = xi - 2.0f * dot * u_i;

  _Float16* zh = pair ? z2h : z1h;
  zh[((size_t)s * B_N + b) * D_N + t] = (_Float16)zi;
}

// ---------------- kernel 3: WMMA GEMM + fixed-shift logsumexp ---------------
// grid = (8 row blocks of 128 rows, 32 samples); 256 threads = 8 waves.
// Wave w owns row tile rt = rb*8+w (A strip kept in registers, K=256).
// All waves share each 16x256 B tile, staged global->LDS with async copies,
// double buffered: async fill of tile j+1 overlaps the WMMA chain of tile j.
// Sched barriers pin the phase order: 16 ds_load_b128 (one clause, one
// s_wait_dscnt) -> 8 back-to-back WMMAs -> packed-VALU exp accumulation.
__global__ __launch_bounds__(256) void mcinfonce_gemm_lse(
    const _Float16* __restrict__ z1h, const _Float16* __restrict__ z2h,
    float* __restrict__ ratio)
{
  const int rb   = blockIdx.x;            // 128-row block
  const int s    = blockIdx.y;            // MC sample
  const int lane = threadIdx.x & 31;
  const int wave = threadIdx.x >> 5;
  const int rt   = rb * 8 + wave;         // this wave's row tile (0..63)

  __shared__ _Float16 shb[2][16 * TPAD_H];   // double-buffered, padded B tile
  __shared__ float sh_diag[8][16];
  __shared__ float sh_rsum[8][16];

  const _Float16* As = z1h + (size_t)s * B_N * D_N;
  const _Float16* Bs = z2h + (size_t)s * B_N * D_N;

  // --- A 16x32 f16 operand strip: lane = M; lanes 0-15 take
  // K 0..15 +32*kk, lanes 16-31 take K 16..31 +32*kk (matches B).
  const _Float16* arow = As + (size_t)(rt * 16 + (lane & 15)) * D_N + ((lane < 16) ? 0 : 16);
  v16h Areg[8];
  #pragma unroll
  for (int kk = 0; kk < 8; ++kk) {
    v8h lo = *(const v8h*)(arow + kk * 32);
    v8h hi = *(const v8h*)(arow + kk * 32 + 8);
    Areg[kk] = __builtin_shufflevector(lo, hi, 0,1,2,3,4,5,6,7,8,9,10,11,12,13,14,15);
  }

  // --- async staging: thread t covers 32B of the tile: row = t>>4, seg = t&15.
  const int srow = threadIdx.x >> 4;
  const int sseg = threadIdx.x & 15;
  const uint32_t lbase = (uint32_t)(srow * TPAD_H * 2 + sseg * 32);   // bytes in tile
  const uint32_t loff0 = (uint32_t)(uintptr_t)(&shb[0][0]) + lbase;
  const uint32_t loff1 = (uint32_t)(uintptr_t)(&shb[1][0]) + lbase;

  #define STAGE_TILE(ct, loff)                                                  \
    do {                                                                        \
      const _Float16* g_ = Bs + (size_t)((ct) * 16 + srow) * D_N + sseg * 16;   \
      uint64_t ga_ = (uint64_t)(uintptr_t)g_;                                   \
      asm volatile("global_load_async_to_lds_b128 %0, %1, off"                  \
                   :: "v"(loff), "v"(ga_) : "memory");                          \
      asm volatile("global_load_async_to_lds_b128 %0, %1, off offset:16"        \
                   :: "v"(loff), "v"(ga_) : "memory");                          \
    } while (0)

  STAGE_TILE(0, loff0);
  asm volatile("s_wait_asynccnt 0x0" ::: "memory");
  __syncthreads();

  float acc[8];           // per-lane column-partial exp sums (reduced after loop)
  #pragma unroll
  for (int r = 0; r < 8; ++r) acc[r] = 0.0f;
  v8f cdiag = {};         // accumulator snapshot of the diagonal tile

  for (int j = 0; j < 64; ++j) {
    if (j + 1 < 64) STAGE_TILE(j + 1, (j & 1) ? loff0 : loff1);  // fill other buffer

    // B 32x16 operand (z2^T): lane = N = tile row; lanes 0-15 K 0..15,
    // lanes 16-31 K 16..31 -> 32 contiguous bytes of the (padded) LDS row.
    const _Float16* sb   = (j & 1) ? &shb[1][0] : &shb[0][0];
    const _Float16* brow = sb + (lane & 15) * TPAD_H + ((lane < 16) ? 0 : 16);

    // phase 1: issue all 16 ds_load_b128 (one clause, one dscnt wait)
    v16h Breg[8];
    #pragma unroll
    for (int kk = 0; kk < 8; ++kk) Breg[kk] = *(const v16h*)(brow + kk * 32);
    SCHED_BARRIER();        // don't sink loads into the WMMA chain

    // phase 2: 8 back-to-back WMMAs
    v8f c = {};
    #pragma unroll
    for (int kk = 0; kk < 8; ++kk) {
      c = __builtin_amdgcn_wmma_f32_16x16x32_f16(
              /*neg_a=*/false, Areg[kk], /*neg_b=*/false, Breg[kk],
              /*c_mod=*/(short)0, c, /*reuse_a=*/false, /*reuse_b=*/false);
    }
    SCHED_BARRIER();        // keep epilogue VALU after the matrix ops

    if (j == rt) cdiag = c;            // uniform scalar branch per wave

    // scores in [-16,16] -> fixed-shift exp; accumulate per lane, reduce later.
    #pragma unroll
    for (int r = 0; r < 8; ++r) acc[r] += __expf(KPOSF * c[r] - KPOSF);

    if (j + 1 < 64) asm volatile("s_wait_asynccnt 0x0" ::: "memory");
    __syncthreads();
  }
  #undef STAGE_TILE

  // ---- epilogue: one 16-lane reduction per row + diagonal extraction ----
  #pragma unroll
  for (int r = 0; r < 8; ++r) {
    float e = acc[r];
    #pragma unroll
    for (int off = 1; off < 16; off <<= 1) e += __shfl_xor(e, off, 16);
    if (lane == 0)  sh_rsum[wave][r]     = e;   // row r of the tile
    if (lane == 16) sh_rsum[wave][r + 8] = e;   // row r+8
    float sc = KPOSF * cdiag[r];
    if (lane == r)      sh_diag[wave][r]     = sc;   // C[M=r][N=r]
    if (lane == r + 24) sh_diag[wave][r + 8] = sc;   // C[M=r+8][N=r+8]
  }
  __syncthreads();
  if (threadIdx.x < 128) {
    const int w = threadIdx.x >> 4, m = threadIdx.x & 15;
    const int row = (rb * 8 + w) * 16 + m;
    float lse = KPOSF + __logf(sh_rsum[w][m]);
    ratio[s * B_N + row] = sh_diag[w][m] - lse + __logf((float)B_N);
  }
}

// ---------------- kernel 4: logsumexp over S, mean over B -------------------
__global__ __launch_bounds__(1024) void mcinfonce_finalize(
    const float* __restrict__ ratio, float* __restrict__ out)
{
  const int b = threadIdx.x;
  float mx = -3.4e38f;
  #pragma unroll
  for (int s = 0; s < S_N; ++s) mx = fmaxf(mx, ratio[s * B_N + b]);
  float sum = 0.0f;
  #pragma unroll
  for (int s = 0; s < S_N; ++s) sum += __expf(ratio[s * B_N + b] - mx);
  float loss = -((mx + __logf(sum)) - __logf((float)S_N));

  __shared__ float sh[32];
  float v = loss;
  #pragma unroll
  for (int off = 16; off > 0; off >>= 1) v += __shfl_xor(v, off, 32);
  if ((threadIdx.x & 31) == 0) sh[threadIdx.x >> 5] = v;
  __syncthreads();
  if (threadIdx.x < 32) {
    float r = sh[threadIdx.x];
    #pragma unroll
    for (int off = 16; off > 0; off >>= 1) r += __shfl_xor(r, off, 32);
    if (threadIdx.x == 0) out[0] = r * (1.0f / (float)B_N);
  }
}

// ---------------- launcher --------------------------------------------------
extern "C" void kernel_launch(void* const* d_in, const int* in_sizes, int n_in,
                              void* d_out, int out_size, void* d_ws, size_t ws_size,
                              hipStream_t stream)
{
  const float* mu1 = (const float*)d_in[0];
  const float* k1r = (const float*)d_in[1];
  const float* mu2 = (const float*)d_in[2];
  const float* k2r = (const float*)d_in[3];
  float* out = (float*)d_out;

  char* ws = (char*)d_ws;
  size_t off = 0;
  float* un  = (float*)(ws + off); off += (size_t)2 * B_N * D_N * sizeof(float);
  float* cst = (float*)(ws + off); off += (size_t)2 * B_N * 4 * sizeof(float);
  _Float16* z1h = (_Float16*)(ws + off); off += (size_t)S_N * B_N * D_N * sizeof(_Float16);
  _Float16* z2h = (_Float16*)(ws + off); off += (size_t)S_N * B_N * D_N * sizeof(_Float16);
  float* ratio = (float*)(ws + off); off += (size_t)S_N * B_N * sizeof(float);
  (void)ws_size; (void)in_sizes; (void)n_in; (void)out_size;

  vmf_prep<<<dim3(B_N, 2), 256, 0, stream>>>(mu1, k1r, mu2, k2r, un, cst);
  vmf_sample<<<dim3(B_N, S_N, 2), 256, 0, stream>>>(un, cst, z1h, z2h);
  mcinfonce_gemm_lse<<<dim3(B_N / 128, S_N), 256, 0, stream>>>(z1h, z2h, ratio);
  mcinfonce_finalize<<<1, 1024, 0, stream>>>(ratio, out);
}
